// GRU_39402029974122
// MI455X (gfx1250) — matrix operand order repaired and checked by reading, not compiled, not run on hardware
//
#include <hip/hip_runtime.h>
#include <hip/hip_bf16.h>
#include <math.h>

// ---------------------------------------------------------------------------
// CDNA5 / gfx1250 WMMA implementation of CNN(3x conv-bn-relu-pool) + GRU + FC.
// Matrix contractions use v_wmma_f32_16x16x32_bf16 (fp32 accumulate); the
// input-projection GEMM stages operands with the Tensor Data Mover
// (tensor_load_to_lds + s_wait_tensorcnt).
// ---------------------------------------------------------------------------

typedef __attribute__((ext_vector_type(16))) __bf16       v16bf;
typedef __attribute__((ext_vector_type(8)))  float        v8f;
typedef __attribute__((ext_vector_type(4)))  unsigned int v4u;
typedef __attribute__((ext_vector_type(8)))  int          v8i;
typedef __attribute__((ext_vector_type(4)))  int          v4i;

__device__ __forceinline__ v8f wmma_bf16(v16bf a, v16bf b, v8f c) {
    // 8 args: (neg_a, A, neg_b, B, c_mod, C, reuse_a, reuse_b)
    return __builtin_amdgcn_wmma_f32_16x16x32_bf16(false, a, false, b,
                                                   (short)0, c, false, false);
}

__device__ __forceinline__ float sigmoidf_(float x) {
    return 1.0f / (1.0f + __expf(-x));
}

// ---------------------------------------------------------------------------
// TDM: 1D tile load of 512 bf16 (1 KB) from global to LDS.
// D# group0: count=1 | lds_addr | global_addr[56:0] | type=2
// D# group1: data_size=1(2B), tensor_dim0=512, tile_dim0=512, rest 0.
// clang-23 toolchain builtin: 6 args (g0, g1, g2, g3, g4, cpol).
// ---------------------------------------------------------------------------
__device__ __forceinline__ void tdm_load_tile512(__bf16* ldsDst, const __bf16* src) {
    unsigned long long ga = (unsigned long long)(const void*)src;
    unsigned ldsOff = (unsigned)(unsigned long long)(void*)ldsDst;
    v4u g0;
    g0[0] = 1u;                                   // count=1, is_restore=0
    g0[1] = ldsOff;                               // lds_addr (bytes)
    g0[2] = (unsigned)ga;                         // global_addr[31:0]
    g0[3] = (unsigned)((ga >> 32) & 0x01ffffffull) | (2u << 30); // addr[56:32], type=2
    v8i g1;
    g1[0] = (int)(1u << 16);                      // data_size = 2 bytes
    g1[1] = (int)(512u << 16);                    // tensor_dim0[15:0] in [31:16]
    g1[2] = 0;                                    // tensor_dim0 hi / tensor_dim1 lo
    g1[3] = (int)(512u << 16);                    // tile_dim0 in [31:16]
    g1[4] = 0; g1[5] = 0; g1[6] = 0; g1[7] = 0;
    v4i z4 = {0, 0, 0, 0};
    v8i z8 = {0, 0, 0, 0, 0, 0, 0, 0};
    __builtin_amdgcn_tensor_load_to_lds(g0, g1, z4, z4, z8, 0);
}

// ---------------------------------------------------------------------------
// Pack a row-major fp32 [M][K] matrix into WMMA A-fragment tiles (bf16).
//   m = mt*16 + (l&15)
//   k = kt*32 + (l<16?0:16) + j
// ---------------------------------------------------------------------------
__global__ void pack_a_tiles(const float* __restrict__ src, int M, int K,
                             int mtiles, int ktiles, __bf16* __restrict__ dst) {
    int idx = blockIdx.x * blockDim.x + threadIdx.x;
    int total = mtiles * ktiles * 512;
    if (idx >= total) return;
    int e    = idx & 511;
    int tile = idx >> 9;
    int lane = e >> 4;
    int j    = e & 15;
    int kt = tile % ktiles;
    int mt = tile / ktiles;
    int m = mt * 16 + (lane & 15);
    int v = j >> 1, h = j & 1;
    int half = (lane < 16) ? 0 : 1;
    int k = kt * 32 + half * 16 + v * 2 + h;
    float val = (m < M && k < K) ? src[(long)m * K + k] : 0.0f;
    dst[idx] = (__bf16)val;
}

// ---------------------------------------------------------------------------
// Pack a row-major fp32 [N][K] matrix into WMMA B-fragment tiles (bf16).
//   n = nt*16 + (l&15) ; k = kt*32 + (l<16?0:16) + j
// ---------------------------------------------------------------------------
__global__ void pack_b_tiles(const float* __restrict__ src, int N, int K,
                             int ntiles, int ktiles, __bf16* __restrict__ dst) {
    int idx = blockIdx.x * blockDim.x + threadIdx.x;
    int total = ntiles * ktiles * 512;
    if (idx >= total) return;
    int e    = idx & 511;
    int tile = idx >> 9;
    int lane = e >> 4;
    int j    = e & 15;
    int kt = tile % ktiles;
    int nt = tile / ktiles;
    int n = nt * 16 + (lane & 15);
    int k = kt * 32 + ((lane < 16) ? 0 : 16) + j;
    float val = (n < N && k < K) ? src[(long)n * K + k] : 0.0f;
    dst[idx] = (__bf16)val;
}

// ---------------------------------------------------------------------------
// Implicit-GEMM conv 7x7 (pad 3) + bias + BN + ReLU + maxpool 2x2.
// One wave per (frame, output y-pair, 16-wide x-strip).
// B fragments for each output row are materialized once into LDS in exact
// WMMA fragment layout (branch-free build), then consumed as contiguous
// 32-byte LDS vector loads in the WMMA K-loop.
// ---------------------------------------------------------------------------
__global__ void __launch_bounds__(32)
conv_bn_pool_wmma(const float* __restrict__ in,
                  int frameStride, int strideC, int strideY, int strideX,
                  int H, int W, int Cin, int Cout,
                  const __bf16* __restrict__ At, int mtiles, int ktiles,
                  const float* __restrict__ bias, const float* __restrict__ gamma,
                  const float* __restrict__ beta, const float* __restrict__ mean,
                  const float* __restrict__ var,
                  float* __restrict__ out, int nyp, int nxs) {
    __shared__ float sIn[2664];                     // [Cin][8][22] + 24-float zero pad
    __shared__ __align__(32) __bf16 sBf[23 * 512];  // B fragments of one row
    __shared__ float sC[2][2][16][16];              // [row][mtile][M][N]
    __shared__ float sScale[32], sShift[32];

    const int ZOFF = 2640;                          // zeroed pad slot base

    int lane = threadIdx.x;
    int bid  = blockIdx.x;
    int xs = bid % nxs; int t = bid / nxs;
    int yp = t % nyp;   int f = t / nyp;
    int x0 = xs * 16, y0 = yp * 2;

    if (lane < Cout) {
        float inv = gamma[lane] * rsqrtf(var[lane] + 1e-5f);
        sScale[lane] = inv;
        sShift[lane] = beta[lane] + (bias[lane] - mean[lane]) * inv;
    } else {
        sScale[lane] = 0.0f;
        sShift[lane] = 0.0f;
    }
    if (lane < 24) sIn[ZOFF + lane] = 0.0f;

    // Stage halo rows [y0-3, y0+4] x cols [x0-3, x0+18] for all channels.
    const float* fin = in + (long)f * frameStride;
    int nstage = Cin * 176;
    for (int idx = lane; idx < nstage; idx += 32) {
        int ci = idx / 176, rem = idx % 176;
        int ry = rem / 22,  rx  = rem % 22;
        int iy = y0 - 3 + ry, ix = x0 - 3 + rx;
        float v = 0.0f;
        if (iy >= 0 && iy < H && ix >= 0 && ix < W)
            v = fin[(long)ci * strideC + (long)iy * strideY + (long)ix * strideX];
        sIn[idx] = v;
    }
    __syncthreads();

    int Kreal = Cin * 49;
    int n = lane & 15;
    int nb = ktiles << 9;

    for (int row = 0; row < 2; ++row) {
        // Build B fragments for this output row (branch-free: OOB -> ZOFF).
        for (int e = lane; e < nb; e += 32) {
            int kt = e >> 9; int r = e & 511;
            int ln = r >> 4; int j = r & 15;
            int nn = ln & 15; int half = ln >> 4;
            int k = (kt << 5) + (half << 4) + j;
            int ci = k / 49; int rem = k - ci * 49;
            int ky = rem / 7; int kx = rem - ky * 7;
            int off = ci * 176 + (row + ky) * 22 + (nn + kx);
            off = (k < Kreal) ? off : ZOFF;
            sBf[e] = (__bf16)sIn[off];
        }
        __syncthreads();
        for (int mt = 0; mt < mtiles; ++mt) {
            v8f c = {};
            for (int kt = 0; kt < ktiles; ++kt) {
                v16bf a = *((const v16bf*)(At + (((long)mt * ktiles + kt) << 9)) + lane);
                v16bf b = *((const v16bf*)(sBf + (kt << 9)) + lane);
                c = wmma_bf16(a, b, c);
            }
#pragma unroll
            for (int r = 0; r < 8; ++r) {
                int ml = (lane < 16) ? r : (8 + r);
                int co = mt * 16 + ml;
                float sc = (co < 32) ? sScale[co] : 0.0f;
                float sf = (co < 32) ? sShift[co] : 0.0f;
                sC[row][mt][ml][n] = fmaxf(c[r] * sc + sf, 0.0f);
            }
        }
        __syncthreads();
    }

    // 2x2 maxpool + store
    int Hp = H >> 1, Wp = W >> 1;
    for (int it = lane; it < mtiles * 128; it += 32) {
        int mt = it >> 7; int r2 = it & 127;
        int ml = r2 >> 3, pj = r2 & 7;
        int co = mt * 16 + ml;
        int px = (x0 >> 1) + pj;
        if (co < Cout && px < Wp) {
            float m0 = fmaxf(sC[0][mt][ml][2 * pj], sC[0][mt][ml][2 * pj + 1]);
            float m1 = fmaxf(sC[1][mt][ml][2 * pj], sC[1][mt][ml][2 * pj + 1]);
            out[(((long)f * Cout + co) * Hp + yp) * Wp + px] = fmaxf(m0, m1);
        }
    }
}

// ---------------------------------------------------------------------------
// gx_all GEMM:  G[frame][m] = sum_k w_ih[m][k] * feats[frame][k] + b_ih[m]
// One wave per 16x16 C-tile; A/B tiles double-buffered into LDS via the
// Tensor Data Mover, pipelined with s_wait_tensorcnt.
// ---------------------------------------------------------------------------
__global__ void __launch_bounds__(32)
gemm_gx_wmma(const __bf16* __restrict__ At, const __bf16* __restrict__ Bt,
             const float* __restrict__ b_ih, float* __restrict__ G,
             int mtiles, int ntiles, int ktiles, int Mreal, int ldG) {
    __shared__ __align__(32) __bf16 sA[2][512];
    __shared__ __align__(32) __bf16 sB[2][512];
    int lane = threadIdx.x;
    int nt = blockIdx.x % ntiles;
    int mt = blockIdx.x / ntiles;
    const __bf16* Ap = At + (long)mt * ktiles * 512;
    const __bf16* Bp = Bt + (long)nt * ktiles * 512;

    tdm_load_tile512(&sA[0][0], Ap);
    tdm_load_tile512(&sB[0][0], Bp);

    v8f c = {};
    for (int kt = 0; kt < ktiles; ++kt) {
        int cur = kt & 1;
        if (kt + 1 < ktiles) {
            tdm_load_tile512(&sA[cur ^ 1][0], Ap + ((long)(kt + 1) << 9));
            tdm_load_tile512(&sB[cur ^ 1][0], Bp + ((long)(kt + 1) << 9));
            __builtin_amdgcn_s_wait_tensorcnt(2);   // current buffer landed
        } else {
            __builtin_amdgcn_s_wait_tensorcnt(0);
        }
        asm volatile("" ::: "memory");
        v16bf a = *((const v16bf*)&sA[cur][0] + lane);
        v16bf b = *((const v16bf*)&sB[cur][0] + lane);
        c = wmma_bf16(a, b, c);
        asm volatile("" ::: "memory");
    }

    int frame = nt * 16 + (lane & 15);
#pragma unroll
    for (int r = 0; r < 8; ++r) {
        int m = mt * 16 + ((lane < 16) ? r : 8 + r);
        if (m < Mreal)
            G[(long)frame * ldG + m] = c[r] + b_ih[m];
    }
}

// ---------------------------------------------------------------------------
// Persistent GRU scan (32 sequential steps) + classifier.
// One workgroup of 32 waves; hidden state lives in LDS as bf16 in B-fragment
// friendly layout (rows 8..15 zero), so each WMMA B operand is one aligned
// 32-byte LDS vector load.
// ---------------------------------------------------------------------------
#define GRU_H   750
#define GRU_G3  2250
#define GRU_MT  141
#define GRU_KT  24
#define GRU_S   32
#define GRU_NB  8

__global__ void __launch_bounds__(1024)
gru_scan_cls(const __bf16* __restrict__ Ahh, const float* __restrict__ G,
             const float* __restrict__ b_hh,
             const float* __restrict__ cls_w, const float* __restrict__ cls_b,
             float* __restrict__ out) {
    __shared__ __align__(32) __bf16 shb[16][768];   // hidden state, rows 8..15 = 0
    __shared__ __bf16 sgh[2256][GRU_NB];            // recurrent pre-activations

    int tid  = threadIdx.x;
    int wave = tid >> 5, lane = tid & 31;

    for (int i = tid; i < 16 * 768; i += 1024)
        (&shb[0][0])[i] = (__bf16)0.0f;
    __syncthreads();

    int n = lane & 15;
    int half16 = (lane < 16) ? 0 : 16;

    for (int s = 0; s < GRU_S; ++s) {
        for (int mt = wave; mt < GRU_MT; mt += 32) {
            v8f c = {};
            const __bf16* Ap = Ahh + (long)mt * GRU_KT * 512;
            for (int kt = 0; kt < GRU_KT; ++kt) {
                if (kt + 1 < GRU_KT)
                    __builtin_prefetch(Ap + ((long)(kt + 1) << 9) + lane * 16, 0, 0);
                v16bf a = *((const v16bf*)(Ap + ((long)kt << 9)) + lane);
                v16bf b = *(const v16bf*)&shb[n][kt * 32 + half16];
                c = wmma_bf16(a, b, c);
            }
            if (n < GRU_NB) {
#pragma unroll
                for (int r = 0; r < 8; ++r) {
                    int m = mt * 16 + ((lane < 16) ? r : 8 + r);
                    if (m < GRU_G3) sgh[m][n] = (__bf16)(c[r] + b_hh[m]);
                }
            }
        }
        __syncthreads();
        for (int idx = tid; idx < GRU_NB * GRU_H; idx += 1024) {
            int b = idx / GRU_H, i = idx - b * GRU_H;
            const float* g = G + (long)(b * GRU_S + s) * GRU_G3;
            float hold = (float)shb[b][i];
            float r  = sigmoidf_(g[i] + (float)sgh[i][b]);
            float z  = sigmoidf_(g[GRU_H + i] + (float)sgh[GRU_H + i][b]);
            float nn = tanhf(g[2 * GRU_H + i] + r * (float)sgh[2 * GRU_H + i][b]);
            shb[b][i] = (__bf16)((1.0f - z) * nn + z * hold);
        }
        __syncthreads();
    }

    // classifier: out[b][c] = h[b] . cls_w[c] + cls_b[c]
    for (int idx = tid; idx < GRU_NB * 11; idx += 1024) {
        int b = idx / 11, cc = idx - b * 11;
        float acc = cls_b[cc];
        const float* wrow = cls_w + (long)cc * GRU_H;
        for (int i = 0; i < GRU_H; ++i) acc += (float)shb[b][i] * wrow[i];
        out[idx] = acc;
    }
}

// ---------------------------------------------------------------------------
// Workspace layout (bytes)
// ---------------------------------------------------------------------------
static constexpr size_t OFF_ACT1  = 0;                          // 256*10*60*80 f32
static constexpr size_t OFF_ACT2  = OFF_ACT1 + 49152000;        // 256*15*30*40 f32
static constexpr size_t OFF_FEATS = OFF_ACT2 + 18432000;        // 256*7500 f32
static constexpr size_t OFF_G     = OFF_FEATS + 7680000;        // 256*2250 f32
static constexpr size_t OFF_A1    = OFF_G + 2304000;            // 1*5*512 bf16
static constexpr size_t OFF_A2    = OFF_A1 + 5120;              // 1*16*512 bf16
static constexpr size_t OFF_A3    = OFF_A2 + 16384;             // 2*23*512 bf16
static constexpr size_t OFF_AHH   = OFF_A3 + 47104;             // 141*24*512 bf16
static constexpr size_t OFF_AIH   = OFF_AHH + 3465216;          // 141*235*512 bf16
static constexpr size_t OFF_BF    = OFF_AIH + 33930240;         // 16*235*512 bf16
// total = OFF_BF + 3850240 = 118,882,304 bytes

extern "C" void kernel_launch(void* const* d_in, const int* in_sizes, int n_in,
                              void* d_out, int out_size, void* d_ws, size_t ws_size,
                              hipStream_t stream) {
    (void)in_sizes; (void)n_in; (void)out_size; (void)ws_size;

    const float* x    = (const float*)d_in[0];
    const float* c1w  = (const float*)d_in[1];
    const float* c1b  = (const float*)d_in[2];
    const float* g1   = (const float*)d_in[3];
    const float* be1  = (const float*)d_in[4];
    const float* m1   = (const float*)d_in[5];
    const float* vv1  = (const float*)d_in[6];
    const float* c2w  = (const float*)d_in[7];
    const float* c2b  = (const float*)d_in[8];
    const float* g2   = (const float*)d_in[9];
    const float* be2  = (const float*)d_in[10];
    const float* m2   = (const float*)d_in[11];
    const float* vv2  = (const float*)d_in[12];
    const float* c3w  = (const float*)d_in[13];
    const float* c3b  = (const float*)d_in[14];
    const float* g3   = (const float*)d_in[15];
    const float* be3  = (const float*)d_in[16];
    const float* m3   = (const float*)d_in[17];
    const float* vv3  = (const float*)d_in[18];
    const float* w_ih = (const float*)d_in[19];
    const float* w_hh = (const float*)d_in[20];
    const float* b_ih = (const float*)d_in[21];
    const float* b_hh = (const float*)d_in[22];
    const float* clsw = (const float*)d_in[23];
    const float* clsb = (const float*)d_in[24];

    char* ws = (char*)d_ws;
    float*  act1  = (float*)(ws + OFF_ACT1);
    float*  act2  = (float*)(ws + OFF_ACT2);
    float*  feats = (float*)(ws + OFF_FEATS);
    float*  Gbuf  = (float*)(ws + OFF_G);
    __bf16* A1    = (__bf16*)(ws + OFF_A1);
    __bf16* A2    = (__bf16*)(ws + OFF_A2);
    __bf16* A3    = (__bf16*)(ws + OFF_A3);
    __bf16* Ahh   = (__bf16*)(ws + OFF_AHH);
    __bf16* Aih   = (__bf16*)(ws + OFF_AIH);
    __bf16* Bf    = (__bf16*)(ws + OFF_BF);

    // Weight packing (bf16 WMMA fragment layout).
    pack_a_tiles<<<(1 * 5 * 512 + 255) / 256, 256, 0, stream>>>(c1w, 10, 147, 1, 5, A1);
    pack_a_tiles<<<(1 * 16 * 512 + 255) / 256, 256, 0, stream>>>(c2w, 15, 490, 1, 16, A2);
    pack_a_tiles<<<(2 * 23 * 512 + 255) / 256, 256, 0, stream>>>(c3w, 25, 735, 2, 23, A3);
    pack_a_tiles<<<(141 * 235 * 512 + 255) / 256, 256, 0, stream>>>(w_ih, 2250, 7500, 141, 235, Aih);
    pack_a_tiles<<<(141 * 24 * 512 + 255) / 256, 256, 0, stream>>>(w_hh, 2250, 750, 141, 24, Ahh);

    // conv1: x [f][120][160][3]  (strideC=1, strideY=480, strideX=3)
    conv_bn_pool_wmma<<<256 * 60 * 10, 32, 0, stream>>>(
        x, 120 * 160 * 3, 1, 160 * 3, 3, 120, 160, 3, 10,
        A1, 1, 5, c1b, g1, be1, m1, vv1, act1, 60, 10);
    // conv2: act1 [f][10][60][80]
    conv_bn_pool_wmma<<<256 * 30 * 5, 32, 0, stream>>>(
        act1, 10 * 60 * 80, 60 * 80, 80, 1, 60, 80, 10, 15,
        A2, 1, 16, c2b, g2, be2, m2, vv2, act2, 30, 5);
    // conv3: act2 [f][15][30][40] -> feats [f][25][15][20] = [256][7500]
    conv_bn_pool_wmma<<<256 * 15 * 3, 32, 0, stream>>>(
        act2, 15 * 30 * 40, 30 * 40, 40, 1, 30, 40, 15, 25,
        A3, 2, 23, c3b, g3, be3, m3, vv3, feats, 15, 3);

    // Input projection GEMM: G[frame][2250] (TDM-staged, WMMA)
    pack_b_tiles<<<(16 * 235 * 512 + 255) / 256, 256, 0, stream>>>(feats, 256, 7500, 16, 235, Bf);
    gemm_gx_wmma<<<141 * 16, 32, 0, stream>>>(Aih, Bf, b_ih, Gbuf, 141, 16, 235, 2250, 2250);

    // Sequential GRU scan + classifier (single persistent workgroup).
    gru_scan_cls<<<1, 1024, 0, stream>>>(Ahh, Gbuf, b_hh, clsw, clsb, (float*)d_out);
}
